// MultiHeadAttention_16776142258374
// MI455X (gfx1250) — matrix operand order repaired and checked by reading, not compiled, not run on hardware
//
#include <hip/hip_runtime.h>
#include <hip/hip_bf16.h>
#include <math.h>

// Problem constants (match reference)
#define B_ 128
#define T_ 256
#define C_ 384
#define H_ 6
#define D_ 64

typedef __bf16 bf16_t;
typedef __attribute__((ext_vector_type(16))) __bf16 v16bf;
typedef __attribute__((ext_vector_type(8)))  float  v8f;

// ---------------------------------------------------------------------------
// D = A(16x32 bf16) x B(32x16 bf16) + C(16x16 f32), wave32 CDNA5 WMMA
// ---------------------------------------------------------------------------
__device__ __forceinline__ v8f wmma_bf16(v16bf a, v16bf b, v8f c) {
  return __builtin_amdgcn_wmma_f32_16x16x32_bf16(
      /*neg_a=*/false, a, /*neg_b=*/false, b,
      /*c_mod=*/(short)0, c, /*reuse_a=*/false, /*reuse_b=*/false);
}

// A-matrix 16x32 (MxK) fragment, contiguous k order (k = 16h + s). Per lane the
// 16 halves cover one 32-byte contiguous run -> SLP vectorizes to 2 x b128 loads.
__device__ __forceinline__ v16bf load_frag_a(const bf16_t* p, int ld, int k0) {
  const int lane = threadIdx.x & 31;
  const int mrow = lane & 15;
  const int koff = (lane < 16) ? 0 : 16;
  v16bf a;
#pragma unroll
  for (int j = 0; j < 8; ++j) {
    const int kb = (j << 1) + koff + k0;
    a[2 * j]     = p[mrow * ld + kb];
    a[2 * j + 1] = p[mrow * ld + kb + 1];
  }
  return a;
}

// B-matrix 32x16 (KxN) fragment from an [N][K] (N-major) source:
// B(K,N) = p[n*ld + K]. Per lane: 16 contiguous bf16 -> 2 x b128 loads.
__device__ __forceinline__ v16bf load_frag_bT(const bf16_t* p, int ld, int k0) {
  const int lane = threadIdx.x & 31;
  const int n    = lane & 15;
  const int koff = ((lane < 16) ? 0 : 16) + k0;
  v16bf b;
#pragma unroll
  for (int j = 0; j < 8; ++j) {
    b[2 * j]     = p[n * ld + koff + 2 * j];
    b[2 * j + 1] = p[n * ld + koff + 2 * j + 1];
  }
  return b;
}

// ---------------------------------------------------------------------------
// Kernel 0: convert fp32 weights to bf16, transposed to [batch][cols][rows]
// (N-major for the bT fragment loader). in: [batch][rows][cols].
// ---------------------------------------------------------------------------
__global__ void cvt_transpose_kernel(const float* __restrict__ in,
                                     bf16_t* __restrict__ out,
                                     int batch, int rows, int cols) {
  int i = blockIdx.x * blockDim.x + threadIdx.x;
  int total = batch * rows * cols;
  if (i >= total) return;
  int c = i % cols;
  int r = (i / cols) % rows;
  int b = i / (cols * rows);
  out[(size_t)b * rows * cols + (size_t)c * rows + r] = (bf16_t)in[i];
}

// ---------------------------------------------------------------------------
// Kernel 1: q,k,v = x @ w{q,k,v}[h]  (bf16 out, layout [b][h][t][d])
// Block: 96 threads (3 waves: q,k,v). Grid: (B*T/16, H).
// Weights pre-transposed bf16 [h][d][c].
// ---------------------------------------------------------------------------
__global__ __launch_bounds__(96) void qkv_kernel(
    const float* __restrict__ x,
    const bf16_t* __restrict__ wqt, const bf16_t* __restrict__ wkt,
    const bf16_t* __restrict__ wvt,
    bf16_t* __restrict__ q, bf16_t* __restrict__ k, bf16_t* __restrict__ v) {
  __shared__ __align__(16) bf16_t x_lds[16 * C_];

  const int tid  = threadIdx.x;
  const int wave = tid >> 5;
  const int lane = tid & 31;
  const int bx   = blockIdx.x;
  const int b    = bx >> 4;        // T/16 == 16 tiles per batch row
  const int qt   = bx & 15;
  const int h    = blockIdx.y;
  const int t0   = qt * 16;

  // Stage 16x384 x-tile to LDS as bf16 (rows contiguous in memory)
  const float* xrow = x + (size_t)(b * T_ + t0) * C_;
  for (int i = tid; i < 16 * C_; i += 96) x_lds[i] = (bf16_t)xrow[i];
  __syncthreads();

  const bf16_t* wt   = (wave == 0) ? wqt : (wave == 1) ? wkt : wvt;
  bf16_t*       dstm = (wave == 0) ? q : (wave == 1) ? k : v;
  const bf16_t* wh   = wt + h * D_ * C_;   // [d][c]

  v8f acc[4] = {};
  for (int kc = 0; kc < C_ / 32; ++kc) {
    v16bf a = load_frag_a(x_lds, C_, kc * 32);
#pragma unroll
    for (int nt = 0; nt < 4; ++nt) {
      v16bf bb = load_frag_bT(wh + nt * 16 * C_, C_, kc * 32);
      acc[nt] = wmma_bf16(a, bb, acc[nt]);
    }
  }

  // Repack C/D fragments through LDS -> coalesced b128 global stores.
  __syncthreads();                              // x_lds no longer needed
  const int n16   = lane & 15;
  const int mBase = (lane < 16) ? 0 : 8;
  bf16_t* st = x_lds + wave * 1024;             // 16x64 bf16 tile per wave
#pragma unroll
  for (int nt = 0; nt < 4; ++nt)
#pragma unroll
    for (int r = 0; r < 8; ++r)
      st[(mBase + r) * D_ + nt * 16 + n16] = (bf16_t)acc[nt][r];
  __builtin_amdgcn_wave_barrier();

  // Tile is 16 consecutive rows x full d-width -> 1024 contiguous elements.
  uint4* gd = (uint4*)(dstm + (size_t)((b * H_ + h) * T_ + t0) * D_);
  const uint4* ls = (const uint4*)st;
#pragma unroll
  for (int i = 0; i < 4; ++i) gd[lane + 32 * i] = ls[lane + 32 * i];
}

// ---------------------------------------------------------------------------
// Kernel 2: causal flash attention per 16-row q-tile.
// Block: 128 threads (4 waves, q-tiles qg*4..qg*4+3). Grid: (4, H, B).
// ---------------------------------------------------------------------------
__global__ __launch_bounds__(128) void attn_kernel(
    const bf16_t* __restrict__ q, const bf16_t* __restrict__ k,
    const bf16_t* __restrict__ v, bf16_t* __restrict__ attn) {
  __shared__ __align__(16) bf16_t k_ch[32 * D_];       // [s][d] (bT-contig in d)
  __shared__ __align__(16) bf16_t v_chT[D_ * 32];      // [d][s] (bT-contig in s)
  __shared__ __align__(16) bf16_t scratch[4][1024];    // P tile + out repack

  const int tid  = threadIdx.x;
  const int wave = tid >> 5;
  const int lane = tid & 31;
  const int qg   = blockIdx.x;            // 0..3
  const int h    = blockIdx.y;
  const int b    = blockIdx.z;
  const int qt   = qg * 4 + wave;         // 0..15
  const int t0   = qt * 16;

  const bf16_t* qb = q + (size_t)((b * H_ + h) * T_) * D_;
  const bf16_t* kb = k + (size_t)((b * H_ + h) * T_) * D_;
  const bf16_t* vb = v + (size_t)((b * H_ + h) * T_) * D_;

  // q-tile fragments (16 x 64) held in registers for the whole pass
  v16bf qa0 = load_frag_a(qb + t0 * D_, D_, 0);
  v16bf qa1 = load_frag_a(qb + t0 * D_, D_, 32);

  const int n16   = lane & 15;
  const int mBase = (lane < 16) ? 0 : 8;

  float m[8], l[8];
  v8f o[4] = {};
#pragma unroll
  for (int r = 0; r < 8; ++r) { m[r] = -__builtin_inff(); l[r] = 0.f; }

  const int myChunks  = (qt + 2) >> 1;            // causal: ceil((t0+16)/32)
  const int blkChunks = (qg * 4 + 5) >> 1;        // max over waves in block

  for (int sc = 0; sc < blkChunks; ++sc) {
    __syncthreads();                              // prev compute done before overwrite
    {   // stage k rows [sc*32, sc*32+32) as-is: 256 x uint4
      const uint4* ks = (const uint4*)(kb + sc * 32 * D_);
      uint4* kd = (uint4*)k_ch;
      for (int i = tid; i < 256; i += 128) kd[i] = ks[i];
      // stage v rows transposed: coalesced b128 reads, b16 scatter to LDS
      const uint4* vs = (const uint4*)(vb + sc * 32 * D_);
      for (int i = tid; i < 256; i += 128) {
        union { uint4 u; bf16_t e[8]; } t; t.u = vs[i];
        const int s  = i >> 3;            // 8 uint4 per 64-wide row
        const int d0 = (i & 7) * 8;
#pragma unroll
        for (int e = 0; e < 8; ++e) v_chT[(d0 + e) * 32 + s] = t.e[e];
      }
    }
    __syncthreads();
    if (sc >= myChunks) continue;                 // barriers stay uniform

    // scores S = q . k^T for 32 s-columns, as two 16x16 tiles
    v8f s0 = {}, s1 = {};
    s0 = wmma_bf16(qa0, load_frag_bT(k_ch, D_, 0), s0);
    s0 = wmma_bf16(qa1, load_frag_bT(k_ch, D_, 32), s0);
    s1 = wmma_bf16(qa0, load_frag_bT(k_ch + 16 * D_, D_, 0), s1);
    s1 = wmma_bf16(qa1, load_frag_bT(k_ch + 16 * D_, D_, 32), s1);

    const int sbase = sc * 32;
    bf16_t* pt = scratch[wave];                   // 16x32 P tile (ld=32)
#pragma unroll
    for (int r = 0; r < 8; ++r) {
      const int trow = t0 + mBase + r;
      float a0 = s0[r] * 0.125f;                  // 1/sqrt(64)
      float a1 = s1[r] * 0.125f;
      if (sbase + n16 > trow)      a0 = -__builtin_inff();   // causal mask
      if (sbase + 16 + n16 > trow) a1 = -__builtin_inff();
      float mx = fmaxf(a0, a1);
#pragma unroll
      for (int off = 1; off < 16; off <<= 1)      // row max across 16 lanes
        mx = fmaxf(mx, __shfl_xor(mx, off, 32));
      const float mn    = fmaxf(m[r], mx);
      const float alpha = __expf(m[r] - mn);
      const float e0    = __expf(a0 - mn);
      const float e1    = __expf(a1 - mn);
      float rs = e0 + e1;
#pragma unroll
      for (int off = 1; off < 16; off <<= 1)      // row sum across 16 lanes
        rs += __shfl_xor(rs, off, 32);
      l[r] = l[r] * alpha + rs;
      m[r] = mn;
      o[0][r] *= alpha; o[1][r] *= alpha; o[2][r] *= alpha; o[3][r] *= alpha;
      pt[(mBase + r) * 32 + n16]      = (bf16_t)e0;
      pt[(mBase + r) * 32 + 16 + n16] = (bf16_t)e1;
    }
    __builtin_amdgcn_wave_barrier();              // LDS write -> read within wave

    // O += P(16x32) @ V(32x64); V fragment contiguous from transposed LDS
    v16bf pa = load_frag_a(pt, 32, 0);
#pragma unroll
    for (int nt = 0; nt < 4; ++nt) {
      v16bf vbb = load_frag_bT(v_chT + nt * 16 * 32, 32, 0);
      o[nt] = wmma_bf16(pa, vbb, o[nt]);
    }
  }

  // Normalize, repack through LDS, coalesced b128 stores.
  bf16_t* st = scratch[wave];                     // 16x64 bf16 tile
#pragma unroll
  for (int nt = 0; nt < 4; ++nt)
#pragma unroll
    for (int r = 0; r < 8; ++r)
      st[(mBase + r) * D_ + nt * 16 + n16] = (bf16_t)(o[nt][r] / l[r]);
  __builtin_amdgcn_wave_barrier();

  const uint4* ls = (const uint4*)st;
#pragma unroll
  for (int i = 0; i < 4; ++i) {
    const int j   = lane + 32 * i;                // 128 uint4, 8 per row
    const int row = j >> 3;
    uint4* gd = (uint4*)(attn + (size_t)(b * T_ + t0 + row) * (H_ * D_) + h * D_);
    gd[j & 7] = ls[j];
  }
}

// ---------------------------------------------------------------------------
// Kernel 3: out = attn @ wp + bp  (fp32 out). Block: 1 wave. Grid: (2048, 6).
// wp pre-transposed bf16 [c_out][hd_in].
// ---------------------------------------------------------------------------
__global__ __launch_bounds__(32) void proj_kernel(
    const bf16_t* __restrict__ attn, const bf16_t* __restrict__ wpt,
    const float* __restrict__ bp, float* __restrict__ out) {
  __shared__ __align__(16) bf16_t a_lds[16 * C_];
  const int tid = threadIdx.x;
  const int mt  = blockIdx.x;            // 16-row tile of (B*T)
  const int n0  = blockIdx.y * 64;

  {   // stage 16x384 bf16 tile: 768 x uint4
    const uint4* src = (const uint4*)(attn + (size_t)mt * 16 * C_);
    uint4* dst = (uint4*)a_lds;
    for (int i = tid; i < 768; i += 32) dst[i] = src[i];
  }
  __syncthreads();

  v8f acc[4] = {};
  for (int kc = 0; kc < C_ / 32; ++kc) {
    v16bf a = load_frag_a(a_lds, C_, kc * 32);
#pragma unroll
    for (int nt = 0; nt < 4; ++nt) {
      v16bf bb = load_frag_bT(wpt + (size_t)(n0 + nt * 16) * C_, C_, kc * 32);
      acc[nt] = wmma_bf16(a, bb, acc[nt]);
    }
  }

  // Bias add + repack through LDS (as f32) -> coalesced b128 stores.
  const int lane  = tid & 31;
  const int n16   = lane & 15;
  const int mBase = (lane < 16) ? 0 : 8;
  float* fs = (float*)a_lds;                      // reuse: 16x64 f32 = 4 KB
#pragma unroll
  for (int nt = 0; nt < 4; ++nt)
#pragma unroll
    for (int r = 0; r < 8; ++r)
      fs[(mBase + r) * 64 + nt * 16 + n16] =
          acc[nt][r] + bp[n0 + nt * 16 + n16];
  __builtin_amdgcn_wave_barrier();

  const uint4* ls = (const uint4*)fs;
#pragma unroll
  for (int i = 0; i < 8; ++i) {
    const int j   = lane + 32 * i;                // 256 uint4, 16 per row
    const int row = j >> 4;
    uint4* gd = (uint4*)(out + (size_t)mt * 16 * C_ + row * C_ + n0);
    gd[j & 15] = ls[j];
  }
}

// ---------------------------------------------------------------------------
extern "C" void kernel_launch(void* const* d_in, const int* in_sizes, int n_in,
                              void* d_out, int out_size, void* d_ws, size_t ws_size,
                              hipStream_t stream) {
  const float* x  = (const float*)d_in[0];
  const float* wq = (const float*)d_in[1];
  const float* wk = (const float*)d_in[2];
  const float* wv = (const float*)d_in[3];
  const float* wp = (const float*)d_in[4];
  const float* bp = (const float*)d_in[5];
  float* out = (float*)d_out;

  const size_t nqkv = (size_t)B_ * H_ * T_ * D_;   // 12.58M elems
  const size_t nw   = (size_t)H_ * C_ * D_;        // 147456 elems per matrix
  bf16_t* qb  = (bf16_t*)d_ws;                     // [b][h][t][d]
  bf16_t* kb  = qb + nqkv;
  bf16_t* vb  = kb + nqkv;
  bf16_t* ab  = vb + nqkv;                         // [b][t][h*D]
  bf16_t* wqt = ab + nqkv;                         // [h][d][c]
  bf16_t* wkt = wqt + nw;
  bf16_t* wvt = wkt + nw;
  bf16_t* wpt = wvt + nw;                          // [c_out][hd_in]

  const int nblk = (int)((nw + 255) / 256);
  cvt_transpose_kernel<<<nblk, 256, 0, stream>>>(wq, wqt, H_, C_, D_);
  cvt_transpose_kernel<<<nblk, 256, 0, stream>>>(wk, wkt, H_, C_, D_);
  cvt_transpose_kernel<<<nblk, 256, 0, stream>>>(wv, wvt, H_, C_, D_);
  cvt_transpose_kernel<<<nblk, 256, 0, stream>>>(wp, wpt, 1, H_ * D_, C_);

  qkv_kernel<<<dim3(B_ * T_ / 16, H_), 96, 0, stream>>>(x, wqt, wkt, wvt, qb, kb, vb);
  attn_kernel<<<dim3(4, H_, B_), 128, 0, stream>>>(qb, kb, vb, ab);
  proj_kernel<<<dim3(B_ * T_ / 16, C_ / 64), 32, 0, stream>>>(ab, wpt, bp, out);
}
